// MultiGenGraspingModel_27401891348686
// MI455X (gfx1250) — matrix operand order repaired and checked by reading, not compiled, not run on hardware
//
#include <hip/hip_runtime.h>

// ---------------------------------------------------------------------------
// VQ nearest-codebook search for MI455X (gfx1250, wave32, WMMA bf16,
// async global->LDS double buffering)
// B=131072, K=1024, D=128
// ---------------------------------------------------------------------------

#define BATCH 131072
#define KCODES 1024
#define DIM 128

typedef __attribute__((ext_vector_type(16))) __bf16 v16bf;
typedef __attribute__((ext_vector_type(8)))  float  v8f;
typedef __attribute__((ext_vector_type(8)))  unsigned int v8u;

// round-to-nearest-even fp32 -> bf16, packed pair into one dword
__device__ __forceinline__ unsigned bf16pk(float lo, float hi) {
    unsigned ul = __builtin_bit_cast(unsigned, lo);
    unsigned uh = __builtin_bit_cast(unsigned, hi);
    ul += 0x7FFFu + ((ul >> 16) & 1u);
    uh += 0x7FFFu + ((uh >> 16) & 1u);
    return (ul >> 16) | (uh & 0xFFFF0000u);
}

// async 16B global -> LDS copy (ASYNCcnt tracked). INST_OFFSET is added to
// BOTH the LDS address and the global address (ISA 08_async_tensor §4.4).
__device__ __forceinline__ void async_cp16(unsigned lds_off,
                                           unsigned long long gaddr, int imm) {
    asm volatile("global_load_async_to_lds_b128 %0, %1, off offset:%2"
                 :: "v"(lds_off), "v"(gaddr), "i"(imm) : "memory");
}

__device__ __forceinline__ void wait_async0() {
    asm volatile("s_wait_asynccnt 0x0" ::: "memory");
}

// ---------------------------------------------------------------------------
// Kernel 1: codebook fp32 -> bf16, per-code squared norm, zero the loss slot.
// 1 wave per code; lane handles 4 contiguous floats. grid=128, block=256.
// ---------------------------------------------------------------------------
__global__ void vq_prep_kernel(const float* __restrict__ cb,
                               unsigned short* __restrict__ cb16,
                               float* __restrict__ e_sq,
                               float* __restrict__ lossOut) {
    int lane = threadIdx.x & 31;
    int wave = threadIdx.x >> 5;
    int code = blockIdx.x * 8 + wave;

    const float4 f = *(const float4*)(cb + (size_t)code * DIM + lane * 4);
    float ss = f.x * f.x + f.y * f.y + f.z * f.z + f.w * f.w;

    uint2 pk;
    pk.x = bf16pk(f.x, f.y);
    pk.y = bf16pk(f.z, f.w);
    *(uint2*)(cb16 + (size_t)code * DIM + lane * 4) = pk;

    #pragma unroll
    for (int m = 1; m < 32; m <<= 1) ss += __shfl_xor(ss, m, 32);
    if (lane == 0) e_sq[code] = ss;

    if (blockIdx.x == 0 && threadIdx.x == 0) *lossOut = 0.0f;
}

// ---------------------------------------------------------------------------
// Kernel 2: main search. block=256 (8 waves), each wave owns 16 rows of z_e.
// Codebook streamed through double-buffered LDS (2 x 32KB) with async copies.
// LDS layout per chunk: row-major 128 codes x 256B, 16B blocks XOR-swizzled
// by (row&15) so the B-tile ds_load_b128 per k-step is bank-conflict-free
// while global/LDS strides stay equal (async offset applies to both).
// grid = BATCH / 128 = 1024.
// ---------------------------------------------------------------------------
#define CHUNK_CODES 128
#define CHUNK_BYTES (CHUNK_CODES * DIM * 2)   // 32768

__device__ __forceinline__ void stage_chunk(const unsigned short* cb16,
                                            unsigned sBase, int ch, int buf,
                                            int tid) {
    // element idx = tid + i*256 (2048 x 16B per chunk); r=idx>>4, c=idx&15
    // LDS byte = buf*CHUNK + r*256 + ((c ^ (r&15))*16); global byte = idx*16.
    // For fixed tid, i advances r by 16 (r&15 const) -> both strides = 4096.
    int r0 = tid >> 4, c0 = tid & 15;
    unsigned lds0 = sBase + buf * CHUNK_BYTES + r0 * 256 +
                    ((unsigned)((c0 ^ r0) & 15) << 4);
    unsigned long long g =
        (unsigned long long)(cb16 + (size_t)ch * CHUNK_CODES * DIM + tid * 8);
    async_cp16(lds0, g, 0 * 4096);
    async_cp16(lds0, g, 1 * 4096);
    async_cp16(lds0, g, 2 * 4096);
    async_cp16(lds0, g, 3 * 4096);
    async_cp16(lds0, g, 4 * 4096);
    async_cp16(lds0, g, 5 * 4096);
    async_cp16(lds0, g, 6 * 4096);
    async_cp16(lds0, g, 7 * 4096);
}

__global__ void __launch_bounds__(256)
vq_main_kernel(const float* __restrict__ z_e,
               const float* __restrict__ cb_f32,
               const unsigned short* __restrict__ cb16,
               const float* __restrict__ e_sq,
               float* __restrict__ zq,
               float* __restrict__ lossOut,
               float* __restrict__ idxOut) {
    __shared__ __align__(16) unsigned char sCB[2 * CHUNK_BYTES];

    const int lane = threadIdx.x & 31;
    const int wave = threadIdx.x >> 5;
    const int sub  = lane & 15;
    const int hi8  = (lane & 16) ? 8 : 0;   // A-layout K-offset / C-row offset
    const int h2   = (lane & 16) ? 2 : 0;   // B-layout 16B-chunk offset
    const int rowBase = (blockIdx.x * 8 + wave) * 16;
    const unsigned sBase = (unsigned)(size_t)(void*)sCB;

    // ---- Kick off chunk 0 into buffer 0 ----
    stage_chunk(cb16, sBase, 0, 0, threadIdx.x);

    // ---- Load 16x128 A tile (bf16, WMMA A layout) + ||z||^2 partials ----
    const float* zrow = z_e + (size_t)(rowBase + sub) * DIM;
    float zsqPart = 0.0f;
    v8u A[4];
    #pragma unroll
    for (int s = 0; s < 4; ++s) {
        const int k0 = s * 32 + 2 * hi8;
        float4 f0 = *(const float4*)(zrow + k0);
        float4 f1 = *(const float4*)(zrow + k0 + 4);
        float4 f2 = *(const float4*)(zrow + k0 + 8);
        float4 f3 = *(const float4*)(zrow + k0 + 12);
        zsqPart += f0.x*f0.x + f0.y*f0.y + f0.z*f0.z + f0.w*f0.w;
        zsqPart += f1.x*f1.x + f1.y*f1.y + f1.z*f1.z + f1.w*f1.w;
        zsqPart += f2.x*f2.x + f2.y*f2.y + f2.z*f2.z + f2.w*f2.w;
        zsqPart += f3.x*f3.x + f3.y*f3.y + f3.z*f3.z + f3.w*f3.w;
        A[s][0] = bf16pk(f0.x, f0.y); A[s][1] = bf16pk(f0.z, f0.w);
        A[s][2] = bf16pk(f1.x, f1.y); A[s][3] = bf16pk(f1.z, f1.w);
        A[s][4] = bf16pk(f2.x, f2.y); A[s][5] = bf16pk(f2.z, f2.w);
        A[s][6] = bf16pk(f3.x, f3.y); A[s][7] = bf16pk(f3.z, f3.w);
    }
    // lane l and l^16 hold the two halves of row (l&15)
    const float zsqFull = zsqPart + __shfl_xor(zsqPart, 16, 32);
    // broadcast to C layout: slot v needs row v (low lanes) / v+8 (high lanes)
    float zsqC[8];
    #pragma unroll
    for (int v = 0; v < 8; ++v) zsqC[v] = __shfl(zsqFull, v + hi8, 32);

    float minv[8];
    int   mini[8];
    #pragma unroll
    for (int v = 0; v < 8; ++v) { minv[v] = 3.4e38f; mini[v] = 0; }

    // ---- Sweep the codebook: compute chunk ch while chunk ch+1 streams ----
    for (int ch = 0; ch < KCODES / CHUNK_CODES; ++ch) {
        const int buf = ch & 1;
        wait_async0();        // my async copies for chunk ch have landed
        __syncthreads();      // everyone's copies landed; buf^1 free to reuse
        if (ch + 1 < KCODES / CHUNK_CODES)
            stage_chunk(cb16, sBase, ch + 1, buf ^ 1, threadIdx.x);

        const unsigned char* bufBase = sCB + buf * CHUNK_BYTES;

        #pragma unroll
        for (int t = 0; t < CHUNK_CODES / 16; ++t) {
            const int codeLocal = t * 16 + sub;
            const int code = ch * CHUNK_CODES + codeLocal;
            const float esq = e_sq[code];
            const unsigned char* rowp = bufBase + codeLocal * 256;

            // B tile (bf16 32x16): lane = column = one codebook row,
            // 16 contiguous bf16 per k-step as two XOR-swizzled 16B blocks.
            v8u Bv[4];
            #pragma unroll
            for (int s = 0; s < 4; ++s) {
                const int cA = ((s * 4 + h2)     ^ sub) << 4;
                const int cB = ((s * 4 + h2 + 1) ^ sub) << 4;
                uint4 b0 = *(const uint4*)(rowp + cA);
                uint4 b1 = *(const uint4*)(rowp + cB);
                Bv[s][0] = b0.x; Bv[s][1] = b0.y; Bv[s][2] = b0.z; Bv[s][3] = b0.w;
                Bv[s][4] = b1.x; Bv[s][5] = b1.y; Bv[s][6] = b1.z; Bv[s][7] = b1.w;
            }

            v8f c = {0.f, 0.f, 0.f, 0.f, 0.f, 0.f, 0.f, 0.f};
            #pragma unroll
            for (int s = 0; s < 4; ++s) {
                c = __builtin_amdgcn_wmma_f32_16x16x32_bf16(
                        false, __builtin_bit_cast(v16bf, A[s]),
                        false, __builtin_bit_cast(v16bf, Bv[s]),
                        (short)0, c, false, false);
            }

            #pragma unroll
            for (int v = 0; v < 8; ++v) {
                float d2 = fmaf(-2.0f, c[v], zsqC[v] + esq);
                bool lt = d2 < minv[v];
                mini[v] = lt ? code : mini[v];
                minv[v] = lt ? d2   : minv[v];
            }
        }
    }

    // ---- Reduce min/argmin across the 16 columns held by the lane group ----
    #pragma unroll
    for (int m = 1; m < 16; m <<= 1) {
        #pragma unroll
        for (int v = 0; v < 8; ++v) {
            float ov = __shfl_xor(minv[v], m, 32);
            int   oi = __shfl_xor(mini[v], m, 32);
            if (ov < minv[v] || (ov == minv[v] && oi < mini[v])) {
                minv[v] = ov; mini[v] = oi;
            }
        }
    }

    // ---- Gather z_q (fp32 codebook), indices, commitment-loss partials ----
    float lossAcc = 0.0f;
    #pragma unroll
    for (int v = 0; v < 8; ++v) {
        const int row  = rowBase + v + hi8;       // low half: v, high: v+8
        const int code = mini[v];                 // uniform within 16-lane group
        const float4* cr = (const float4*)(cb_f32 + (size_t)code * DIM + sub * 8);
        const float4* zr = (const float4*)(z_e   + (size_t)row  * DIM + sub * 8);
        float4 c0 = cr[0], c1 = cr[1];
        float4 z0 = zr[0], z1 = zr[1];
        float4* o = (float4*)(zq + (size_t)row * DIM + sub * 8);
        o[0] = c0; o[1] = c1;
        float d;
        d = c0.x - z0.x; lossAcc += d * d;
        d = c0.y - z0.y; lossAcc += d * d;
        d = c0.z - z0.z; lossAcc += d * d;
        d = c0.w - z0.w; lossAcc += d * d;
        d = c1.x - z1.x; lossAcc += d * d;
        d = c1.y - z1.y; lossAcc += d * d;
        d = c1.z - z1.z; lossAcc += d * d;
        d = c1.w - z1.w; lossAcc += d * d;
        if (sub == 0) idxOut[row] = (float)code;
    }

    #pragma unroll
    for (int m = 1; m < 32; m <<= 1) lossAcc += __shfl_xor(lossAcc, m, 32);
    if (lane == 0)
        atomicAdd(lossOut, lossAcc * (1.0f / ((float)BATCH * (float)DIM)));
}

// ---------------------------------------------------------------------------
// Launch
// ---------------------------------------------------------------------------
extern "C" void kernel_launch(void* const* d_in, const int* in_sizes, int n_in,
                              void* d_out, int out_size, void* d_ws, size_t ws_size,
                              hipStream_t stream) {
    const float* z_e = (const float*)d_in[0];   // [B, D] fp32
    const float* cb  = (const float*)d_in[1];   // [K, D] fp32

    unsigned short* cb16 = (unsigned short*)d_ws;                   // 256 KB
    float* e_sq = (float*)((char*)d_ws + (size_t)KCODES * DIM * 2); // 4 KB

    float* zq      = (float*)d_out;                       // B*D floats
    float* lossOut = zq + (size_t)BATCH * DIM;            // 1 float
    float* idxOut  = lossOut + 1;                         // B floats (indices)

    vq_prep_kernel<<<KCODES / 8, 256, 0, stream>>>(cb, cb16, e_sq, lossOut);
    vq_main_kernel<<<BATCH / 128, 256, 0, stream>>>(z_e, cb, cb16, e_sq,
                                                    zq, lossOut, idxOut);
}
